// c_Attention_27152783245798
// MI455X (gfx1250) — matrix operand order repaired and checked by reading, not compiled, not run on hardware
//
#include <hip/hip_runtime.h>
#include <math.h>

// Problem constants (match reference)
#define B_    4
#define N_    1024
#define DIM_  512
#define H_    8
#define DH_   64
#define INNER_ 512

typedef _Float16 half_t;
typedef __attribute__((ext_vector_type(16))) _Float16 v16h;
typedef __attribute__((ext_vector_type(8)))  _Float16 v8h;
typedef __attribute__((ext_vector_type(8)))  float    v8f;
typedef __attribute__((__vector_size__(4 * sizeof(int)))) int i32x4;

// CDNA5 async global->LDS path (guarded: falls back to direct global loads).
#if defined(__AMDGCN__) && __has_builtin(__builtin_amdgcn_global_load_async_to_lds_b128) && __has_builtin(__builtin_amdgcn_s_wait_asynccnt)
#define USE_ASYNC_LDS 1
#else
#define USE_ASYNC_LDS 0
#endif

// ---------------------------------------------------------------------------
// WMMA wrapper: D = A(16x32 f16) * B(32x16 f16) + C(16x16 f32)
// ---------------------------------------------------------------------------
__device__ __forceinline__ v8f wmma16(v16h a, v16h b, v8f c) {
  return __builtin_amdgcn_wmma_f32_16x16x32_f16(
      /*neg_a=*/false, a, /*neg_b=*/false, b,
      /*c_mod=*/(short)0, c, /*reuse_a=*/false, /*reuse_b=*/false);
}

// ---------------------------------------------------------------------------
// Fragment loaders per CDNA5 ISA §7.12.2 layouts (16-bit A 16x32, B 32x16).
// ---------------------------------------------------------------------------
__device__ __forceinline__ v16h frag_a_f16(const half_t* __restrict__ p, int ld,
                                           int row0, int k0, int lane) {
  const half_t* q = p + (size_t)(row0 + (lane & 15)) * ld + k0 + ((lane & 16) ? 8 : 0);
  v8h lo = *(const v8h*)q;
  v8h hi = *(const v8h*)(q + 16);
  v16h r;
#pragma unroll
  for (int i = 0; i < 8; ++i) { r[i] = lo[i]; r[i + 8] = hi[i]; }
  return r;
}

__device__ __forceinline__ v16h frag_b_f16(const half_t* __restrict__ p, int ld,
                                           int n0, int k0, int lane) {
  const half_t* q = p + (size_t)(n0 + (lane & 15)) * ld + k0 + ((lane & 16) ? 8 : 0);
  v8h lo = *(const v8h*)q;
  v8h hi = *(const v8h*)(q + 16);
  v16h r;
#pragma unroll
  for (int i = 0; i < 8; ++i) { r[i] = lo[i]; r[i + 8] = hi[i]; }
  return r;
}

#if USE_ASYNC_LDS
// Stage one contiguous 2048-byte K-tile (16 rows x 64 halves) into LDS with
// async copies: 4 x (32 lanes x 16B) = 2048B, tracked by ASYNCcnt.
// Builtin signature (from compiler diagnostics): (i32x4* AS1, i32x4* AS3, Iimm, Iimm)
__device__ __forceinline__ void stage_tile_async(const half_t* g, half_t* l, int lane) {
  half_t* gnc = const_cast<half_t*>(g) + lane * 8;  // 16 bytes per lane
  half_t* ls  = l + lane * 8;
  auto gp = (__attribute__((address_space(1))) i32x4*)gnc;
  auto lp = (__attribute__((address_space(3))) i32x4*)ls;
  __builtin_amdgcn_global_load_async_to_lds_b128(gp, lp, 0, 0);
  __builtin_amdgcn_global_load_async_to_lds_b128(gp, lp, 512, 0);
  __builtin_amdgcn_global_load_async_to_lds_b128(gp, lp, 1024, 0);
  __builtin_amdgcn_global_load_async_to_lds_b128(gp, lp, 1536, 0);
}

// B-fragment gather from a staged 16x64 LDS tile (tile-local rows), via
// addrspace(3) pointers so ds_load_b128 is emitted.
__device__ __forceinline__ v16h frag_b_lds(const half_t* base, int k0, int lane) {
  const half_t* q = base + (lane & 15) * DH_ + k0 + ((lane & 16) ? 8 : 0);
  auto p = (__attribute__((address_space(3))) const v8h*)q;
  v8h lo = p[0];
  v8h hi = p[2];
  v16h r;
#pragma unroll
  for (int i = 0; i < 8; ++i) { r[i] = lo[i]; r[i + 8] = hi[i]; }
  return r;
}
#endif

// ---------------------------------------------------------------------------
// Kernel 0: f32 -> f16 conversion (x and the four weight matrices), done once
// so the GEMM inner loops carry no v_cvt work and half the load bytes.
// n must be a multiple of 8.
// ---------------------------------------------------------------------------
__global__ __launch_bounds__(256) void cvt_kernel(const float* __restrict__ src,
                                                  half_t* __restrict__ dst, int n) {
  int i = (blockIdx.x * 256 + threadIdx.x) * 8;
  if (i >= n) return;
  float4 a = *(const float4*)(src + i);
  float4 b = *(const float4*)(src + i + 4);
  v8h o;
  o[0] = (half_t)a.x; o[1] = (half_t)a.y; o[2] = (half_t)a.z; o[3] = (half_t)a.w;
  o[4] = (half_t)b.x; o[5] = (half_t)b.y; o[6] = (half_t)b.z; o[7] = (half_t)b.w;
  *(v8h*)(dst + i) = o;
}

// ---------------------------------------------------------------------------
// Kernel 1: QKV projection. grid (32 token-tiles, 8 heads, 3 matrices),
// block 256 = 8 waves; wave = 16 tokens x full head (64 outputs).
// Writes Qh/Kh f16 [b,h,n,dh], V transposed f16 [b,h,dh,n], f32 q/k norms
// computed from the f32 accumulators.
// ---------------------------------------------------------------------------
__global__ __launch_bounds__(256) void qkv_kernel(
    const half_t* __restrict__ xh,
    const half_t* __restrict__ Wq, const float* __restrict__ bq,
    const half_t* __restrict__ Wk, const float* __restrict__ bk,
    const half_t* __restrict__ Wv, const float* __restrict__ bv,
    half_t* __restrict__ Qh, half_t* __restrict__ Kh, half_t* __restrict__ Vt,
    float* __restrict__ qn, float* __restrict__ kn) {
  const int lane = threadIdx.x & 31;
  const int wave = threadIdx.x >> 5;
  const int sel  = blockIdx.z;                  // 0=q 1=k 2=v
  const int h    = blockIdx.y;
  const int t0   = blockIdx.x * 128 + wave * 16; // global token row base
  const int b    = t0 >> 10;
  const int n0   = t0 & (N_ - 1);
  const size_t bh = (size_t)b * H_ + h;

  const half_t* W   = (sel == 0) ? Wq : (sel == 1) ? Wk : Wv;
  const float* bias = (sel == 0) ? bq : (sel == 1) ? bk : bv;

  v8f acc[4] = {};

  for (int k0 = 0; k0 < DIM_; k0 += 32) {
    v16h a = frag_a_f16(xh, DIM_, t0, k0, lane);
#pragma unroll
    for (int t = 0; t < 4; ++t) {
      v16h bf = frag_b_f16(W, DIM_, h * 64 + t * 16, k0, lane);
      acc[t] = wmma16(a, bf, acc[t]);
    }
  }

  const int nb = lane & 15;
  const int hi = (lane >> 4) & 1;

#pragma unroll
  for (int t = 0; t < 4; ++t) {
    float bb = bias[h * 64 + t * 16 + nb];
#pragma unroll
    for (int g = 0; g < 8; ++g) acc[t][g] += bb;
  }

  if (sel == 2) {
    // V transposed: Vt[(bh*64 + dh)*N + n]
#pragma unroll
    for (int t = 0; t < 4; ++t)
#pragma unroll
      for (int g = 0; g < 8; ++g)
        Vt[(bh * 64 + t * 16 + nb) * (size_t)N_ + (n0 + g + hi * 8)] =
            (half_t)acc[t][g];
  } else {
    half_t* O   = (sel == 0) ? Qh : Kh;
    float*  nrm = (sel == 0) ? qn : kn;
#pragma unroll
    for (int t = 0; t < 4; ++t)
#pragma unroll
      for (int g = 0; g < 8; ++g)
        O[(bh * N_ + n0 + g + hi * 8) * (size_t)DH_ + t * 16 + nb] =
            (half_t)acc[t][g];
    // row-norm from f32 accumulators: reduce across 16-lane half groups
#pragma unroll
    for (int g = 0; g < 8; ++g) {
      float s = 0.f;
#pragma unroll
      for (int t = 0; t < 4; ++t) s += acc[t][g] * acc[t][g];
      s += __shfl_xor(s, 1, 32);
      s += __shfl_xor(s, 2, 32);
      s += __shfl_xor(s, 4, 32);
      s += __shfl_xor(s, 8, 32);
      if (nb == 0) nrm[bh * N_ + n0 + g + hi * 8] = sqrtf(s);
    }
  }
}

// ---------------------------------------------------------------------------
// Kernel 2: fused scores + masked softmax + contrastive loss + attn@V.
// One wave (32 threads) per (b, h, 16-row i-tile).
// Dynamic LDS: 16x1024 f32 score strip + 2x2KB K-tile staging buffers.
// ---------------------------------------------------------------------------
__global__ __launch_bounds__(32) void attn_kernel(
    const half_t* __restrict__ Qh, const half_t* __restrict__ Kh,
    const half_t* __restrict__ Vt,
    const float* __restrict__ qn, const float* __restrict__ kn,
    const float* __restrict__ mask,
    float* __restrict__ attn, half_t* __restrict__ Oc,
    float* __restrict__ dcl) {
  extern __shared__ float sc[];  // [16][1024] scores, then K staging

  const int lane = threadIdx.x;
  const int bid  = blockIdx.x;
  const int it   = bid & 63;
  const int h    = (bid >> 6) & 7;
  const int b    = bid >> 9;
  const size_t bh = (size_t)b * H_ + h;
  const int i0   = it * 16;

  const half_t* Q   = Qh + bh * (size_t)(N_ * DH_);
  const half_t* K   = Kh + bh * (size_t)(N_ * DH_);
  const half_t* V   = Vt + bh * (size_t)(N_ * DH_);   // [dh][n]
  const float*  knr = kn + bh * N_;
  const float*  qnr = qn + bh * N_;
  const float*  mrow0 = mask + (size_t)b * N_ * N_;

  const int nb = lane & 15;
  const int hi = lane >> 4;

  // ---- Phase A: scores = q @ k^T into LDS ----
  // Q fragments are invariant over j-tiles: hoist them.
  const v16h qa0 = frag_a_f16(Q, DH_, i0, 0, lane);
  const v16h qa1 = frag_a_f16(Q, DH_, i0, 32, lane);

#if USE_ASYNC_LDS
  half_t* kst = (half_t*)(sc + 16 * N_);       // 2 x (16*64) halves
  stage_tile_async(K, kst, lane);              // prefetch tile 0 -> buf 0
  for (int jt = 0; jt < 64; ++jt) {
    const int buf = jt & 1;
    if (jt + 1 < 64) {
      stage_tile_async(K + (size_t)(jt + 1) * 16 * DH_,
                       kst + (buf ^ 1) * (16 * DH_), lane);
      __builtin_amdgcn_s_wait_asynccnt(4);     // tile jt done, jt+1 in flight
    } else {
      __builtin_amdgcn_s_wait_asynccnt(0);     // drain for the last tile
    }
    const half_t* kb = kst + buf * (16 * DH_);
    v8f s = {};
    s = wmma16(qa0, frag_b_lds(kb, 0, lane), s);
    s = wmma16(qa1, frag_b_lds(kb, 32, lane), s);
#pragma unroll
    for (int g = 0; g < 8; ++g)
      sc[(g + hi * 8) * N_ + jt * 16 + nb] = s[g];
  }
#else
  for (int jt = 0; jt < 64; ++jt) {
    v8f s = {};
    s = wmma16(qa0, frag_b_f16(K, DH_, jt * 16, 0, lane), s);
    s = wmma16(qa1, frag_b_f16(K, DH_, jt * 16, 32, lane), s);
#pragma unroll
    for (int g = 0; g < 8; ++g)
      sc[(g + hi * 8) * N_ + jt * 16 + nb] = s[g];
  }
#endif

  const float scale   = 0.125f;                  // 1/sqrt(DH)
  const float invTemp = 10.0f;                   // 1/TEMP
  const float LNEG    = -23.025850929940457f;    // -log(1e10)

  float clTot   = 0.f;
  float regLane = 0.f;

  // ---- Phase B: per-row softmax + dcl terms, attn to global ----
  for (int r = 0; r < 16; ++r) {
    const int ig = i0 + r;
    const float qni = qnr[ig];
    const float* mrow = mrow0 + (size_t)ig * N_;
    __builtin_prefetch(mrow + lane, 0, 0);

    float zmax = -3.4e38f, cmax = -3.4e38f;
    for (int j = lane; j < N_; j += 32) {
      float s = sc[r * N_ + j];
      float m = mrow[j];
      float z = s * m + (m == 0.f ? LNEG : 0.f);
      float c = s / (qni * knr[j]) * invTemp;
      zmax = fmaxf(zmax, z);
      cmax = fmaxf(cmax, c);
      if (h == 0) regLane += fabsf((j == ig ? 1.f : 0.f) - m);
    }
#pragma unroll
    for (int mm = 16; mm >= 1; mm >>= 1) {
      zmax = fmaxf(zmax, __shfl_xor(zmax, mm, 32));
      cmax = fmaxf(cmax, __shfl_xor(cmax, mm, 32));
    }

    float sz = 0.f, sa = 0.f, sp = 0.f;
    for (int j = lane; j < N_; j += 32) {
      float s = sc[r * N_ + j];
      float m = mrow[j];
      float z = s * m + (m == 0.f ? LNEG : 0.f);
      float c = s / (qni * knr[j]) * invTemp;
      float ez = __expf(scale * (z - zmax));
      float ec = __expf(c - cmax);
      sz += ez; sa += ec; sp += ec * m;
      sc[r * N_ + j] = ez;   // overwrite with unnormalized softmax numerator
    }
#pragma unroll
    for (int mm = 16; mm >= 1; mm >>= 1) {
      sz += __shfl_xor(sz, mm, 32);
      sa += __shfl_xor(sa, mm, 32);
      sp += __shfl_xor(sp, mm, 32);
    }

    const float inv = 1.f / sz;
    float* arow = attn + (bh * N_ + ig) * (size_t)N_;
    for (int j = lane; j < N_; j += 32) {
      float a = sc[r * N_ + j] * inv;
      sc[r * N_ + j] = a;
      arow[j] = a;
    }
    clTot += __logf(sa) - __logf(sp);   // -log(pos/allsum)
  }

  // ---- Phase C: out_tile = attn @ V (A-frags gathered from LDS) ----
  v8f oc[4] = {};
  for (int j0 = 0; j0 < N_; j0 += 32) {
    const int rr = lane & 15;
    const int kb = j0 + ((lane & 16) ? 8 : 0);
    v16h a;
#pragma unroll
    for (int i = 0; i < 8; ++i) {
      a[i]     = (half_t)sc[rr * N_ + kb + i];
      a[i + 8] = (half_t)sc[rr * N_ + kb + 16 + i];
    }
#pragma unroll
    for (int t = 0; t < 4; ++t) {
      v16h bf = frag_b_f16(V, N_, t * 16, j0, lane);
      oc[t] = wmma16(a, bf, oc[t]);
    }
  }
#pragma unroll
  for (int t = 0; t < 4; ++t)
#pragma unroll
    for (int g = 0; g < 8; ++g)
      Oc[((size_t)b * N_ + i0 + g + hi * 8) * INNER_ + h * 64 + t * 16 + nb] =
          (half_t)oc[t][g];

  // ---- loss accumulation ----
  if (h == 0) {
#pragma unroll
    for (int mm = 16; mm >= 1; mm >>= 1)
      regLane += __shfl_xor(regLane, mm, 32);
    if (lane == 0)
      atomicAdd(dcl, regLane * (0.3f / ((float)B_ * N_ * (N_ - 1))));
  }
  if (lane == 0)
    atomicAdd(dcl, clTot * (1.f / ((float)B_ * H_ * N_)));
}

// ---------------------------------------------------------------------------
// Kernel 3: output projection out = Oc @ Wo^T + bo. grid (32, 8), block 256.
// ---------------------------------------------------------------------------
__global__ __launch_bounds__(256) void proj_kernel(
    const half_t* __restrict__ Oc, const half_t* __restrict__ Wo,
    const float* __restrict__ bo, float* __restrict__ out) {
  const int lane = threadIdx.x & 31;
  const int wave = threadIdx.x >> 5;
  const int t0 = blockIdx.x * 128 + wave * 16;
  const int o0 = blockIdx.y * 64;

  v8f acc[4] = {};
  for (int k0 = 0; k0 < INNER_; k0 += 32) {
    v16h a = frag_a_f16(Oc, INNER_, t0, k0, lane);
#pragma unroll
    for (int t = 0; t < 4; ++t) {
      v16h bf = frag_b_f16(Wo, INNER_, o0 + t * 16, k0, lane);
      acc[t] = wmma16(a, bf, acc[t]);
    }
  }

  const int nb = lane & 15;
  const int hi = (lane >> 4) & 1;
#pragma unroll
  for (int t = 0; t < 4; ++t) {
    float bb = bo[o0 + t * 16 + nb];
#pragma unroll
    for (int g = 0; g < 8; ++g)
      out[(size_t)(t0 + g + hi * 8) * DIM_ + o0 + t * 16 + nb] = acc[t][g] + bb;
  }
}

// ---------------------------------------------------------------------------
extern "C" void kernel_launch(void* const* d_in, const int* in_sizes, int n_in,
                              void* d_out, int out_size, void* d_ws, size_t ws_size,
                              hipStream_t stream) {
  (void)in_sizes; (void)n_in; (void)out_size; (void)ws_size;

  const float* x    = (const float*)d_in[0];
  const float* mask = (const float*)d_in[1];
  const float* Wq   = (const float*)d_in[2];
  const float* bq   = (const float*)d_in[3];
  const float* Wk   = (const float*)d_in[4];
  const float* bk   = (const float*)d_in[5];
  const float* Wv   = (const float*)d_in[6];
  const float* bv   = (const float*)d_in[7];
  const float* Wo   = (const float*)d_in[8];
  const float* bo   = (const float*)d_in[9];

  char* ws = (char*)d_ws;
  const size_t MB = 1024 * 1024;
  half_t* Qh  = (half_t*)(ws);                    // 4 MB
  half_t* Kh  = (half_t*)(ws + 4 * MB);           // 4 MB
  half_t* Vt  = (half_t*)(ws + 8 * MB);           // 4 MB
  half_t* Oc  = (half_t*)(ws + 12 * MB);          // 4 MB
  half_t* xh  = (half_t*)(ws + 16 * MB);          // 4 MB
  half_t* Whq = (half_t*)(ws + 20 * MB);          // 512 KB
  half_t* Whk = (half_t*)(ws + 20 * MB + 512 * 1024);
  half_t* Whv = (half_t*)(ws + 21 * MB);
  half_t* Who = (half_t*)(ws + 21 * MB + 512 * 1024);
  float*  qn  = (float*) (ws + 22 * MB);          // 128 KB
  float*  kn  = (float*) (ws + 22 * MB + 128 * 1024);

  float* out  = (float*)d_out;
  float* attn = out + (size_t)B_ * N_ * DIM_;
  float* dcl  = attn + (size_t)B_ * H_ * N_ * N_;

  (void)hipMemsetAsync(dcl, 0, sizeof(float), stream);

  // f32 -> f16 prep (x: 2M elems; each W: 256K elems)
  cvt_kernel<<<dim3((B_ * N_ * DIM_) / 2048), 256, 0, stream>>>(x, xh, B_ * N_ * DIM_);
  cvt_kernel<<<dim3((INNER_ * DIM_) / 2048), 256, 0, stream>>>(Wq, Whq, INNER_ * DIM_);
  cvt_kernel<<<dim3((INNER_ * DIM_) / 2048), 256, 0, stream>>>(Wk, Whk, INNER_ * DIM_);
  cvt_kernel<<<dim3((INNER_ * DIM_) / 2048), 256, 0, stream>>>(Wv, Whv, INNER_ * DIM_);
  cvt_kernel<<<dim3((DIM_ * INNER_) / 2048), 256, 0, stream>>>(Wo, Who, DIM_ * INNER_);

  qkv_kernel<<<dim3(32, 8, 3), 256, 0, stream>>>(
      xh, Whq, bq, Whk, bk, Whv, bv, Qh, Kh, Vt, qn, kn);

  // 64 KB score strip + 4 KB K-tile double buffer
  attn_kernel<<<dim3(B_ * H_ * (N_ / 16)), 32,
                16 * N_ * sizeof(float) + 4096, stream>>>(
      Qh, Kh, Vt, qn, kn, mask, attn, Oc, dcl);

  proj_kernel<<<dim3(32, 8), 256, 0, stream>>>(Oc, Who, bo, out);
}
